// Decoder_25220047962314
// MI455X (gfx1250) — matrix operand order repaired and checked by reading, not compiled, not run on hardware
//
#include <hip/hip_runtime.h>
#include <hip/hip_bf16.h>

typedef _Float16 v16h __attribute__((ext_vector_type(16)));
typedef float    v8f  __attribute__((ext_vector_type(8)));
typedef int      v4i  __attribute__((ext_vector_type(4)));

#define D 256
#define TOK 4096   // B*S = 64*64
#define BATCH 64
#define SEQ 64
#define NSPLIT 4   // candidate-dimension split of the denominator GEMM

// CDNA5 async global->LDS path (ASYNCcnt), if the toolchain exposes it.
#if defined(__has_builtin)
#  if __has_builtin(__builtin_amdgcn_global_load_async_to_lds_b128) && \
      __has_builtin(__builtin_amdgcn_s_wait_asynccnt)
#    define USE_ASYNC_LDS 1
#  endif
#endif
#ifndef USE_ASYNC_LDS
#  define USE_ASYNC_LDS 0
#endif

#if USE_ASYNC_LDS
typedef __attribute__((address_space(1))) v4i* as1_v4i_p;
typedef __attribute__((address_space(3))) v4i* as3_v4i_p;
#endif

// ---------------- fragment loaders (contiguous-K layouts, wave32) -------------
// A 16x32 f16 (MxK): lanes 0-15 -> M=lane, K = 0..15
//                    lanes16-31 -> M=lane-16, K = 16..31
__device__ inline v16h load_a_frag(const _Float16* base, int row, int kb, int lane) {
    const _Float16* p = base + (size_t)row * D + kb + ((lane >> 4) << 4);
    union { uint4 q[2]; v16h v; } u;
    u.q[0] = *reinterpret_cast<const uint4*>(p);        // K rel 0..7
    u.q[1] = *reinterpret_cast<const uint4*>(p + 8);    // K rel 8..15
    return u.v;
}
// B 32x16 f16 (KxN), element (k,n) = SRCROW[n][k] (candidate-row-major source):
// lanes 0-15 -> N=lane, K=0..15 ; lanes 16-31 -> N=lane-16, K=16..31
__device__ inline v16h load_b_frag(const _Float16* base, int row, int kb, int lane) {
    const _Float16* p = base + (size_t)row * D + kb + ((lane >> 4) << 4);
    union { uint4 q[2]; v16h v; } u;
    u.q[0] = *reinterpret_cast<const uint4*>(p);        // K rel 0..7
    u.q[1] = *reinterpret_cast<const uint4*>(p + 8);    // K rel 8..15
    return u.v;
}
// B fragment from LDS tile staged in fragment order:
// entry (ks, lane) = 16 contiguous halves the lane consumes at K-step ks.
__device__ inline v16h load_b_lds(const _Float16* buf, int ks, int lane) {
    const uint4* p = reinterpret_cast<const uint4*>(buf + (size_t)(ks * 32 + lane) * 16);
    union { uint4 q[2]; v16h v; } u;
    u.q[0] = p[0]; u.q[1] = p[1];
    return u.v;
}

__device__ inline float seg16_reduce(float v) {
    v += __shfl_xor(v, 1, 32);
    v += __shfl_xor(v, 2, 32);
    v += __shfl_xor(v, 4, 32);
    v += __shfl_xor(v, 8, 32);
    return v;
}

// ---------------- small utility kernels -------------------------------------
__global__ void zero_f32_kernel(float* p, int n) {
    int i = blockIdx.x * blockDim.x + threadIdx.x;
    if (i < n) p[i] = 0.0f;
}

__global__ void cvt_f16_kernel(const float* __restrict__ in, _Float16* __restrict__ out) {
    int t = blockIdx.x, d = threadIdx.x;
    out[(size_t)t * D + d] = (_Float16)in[(size_t)t * D + d];
}

__global__ void gather_rows_f16_kernel(const float* __restrict__ W,
                                       const int* __restrict__ idx,
                                       _Float16* __restrict__ out) {
    int t = blockIdx.x, d = threadIdx.x;
    int row = idx[t];
    out[(size_t)t * D + d] = (_Float16)W[(size_t)row * D + d];
}

__global__ void gather_cand_f16_kernel(const float* __restrict__ W,
                                       const int* __restrict__ pos,
                                       const int* __restrict__ neg,
                                       int P, int Ncand,
                                       _Float16* __restrict__ G) {
    int c = blockIdx.x, d = threadIdx.x;
    float v = 0.0f;
    if (c < Ncand) {
        int row = (c < P) ? pos[c] : neg[c - P];
        v = W[(size_t)row * D + d];
    }
    G[(size_t)c * D + d] = (_Float16)v;   // pad rows -> 0 (masked in epilogue)
}

// ---------------- B-tile staging (fragment order, 32B per thread) -----------
__device__ inline void stage_gload(const _Float16* __restrict__ G, int nt, int tid,
                                   uint4* q0, uint4* q1) {
    const int ks = tid >> 5, l = tid & 31;
    const _Float16* p = G + (size_t)(nt * 16 + (l & 15)) * D + ks * 32 + ((l >> 4) << 4);
    *q0 = *reinterpret_cast<const uint4*>(p);
    *q1 = *reinterpret_cast<const uint4*>(p + 8);
}
__device__ inline void stage_sstore(_Float16* buf, int tid, uint4 q0, uint4 q1) {
    uint4* s = reinterpret_cast<uint4*>(buf) + tid * 2;
    s[0] = q0; s[1] = q1;
}

#if USE_ASYNC_LDS
// Direct global->LDS DMA (GLOBAL_LOAD_ASYNC_TO_LDS_B128, ASYNCcnt-tracked):
// no VGPR staging, copy latency hidden under the WMMA burst of the live tile.
__device__ inline void stage_async(const _Float16* __restrict__ G, int nt, int tid,
                                   _Float16* buf) {
    const int ks = tid >> 5, l = tid & 31;
    const _Float16* gp = G + (size_t)(nt * 16 + (l & 15)) * D + ks * 32 + ((l >> 4) << 4);
    _Float16* sp = buf + (size_t)tid * 16;
    __builtin_amdgcn_global_load_async_to_lds_b128(
        (as1_v4i_p)gp, (as3_v4i_p)sp, 0, 0);
    __builtin_amdgcn_global_load_async_to_lds_b128(
        (as1_v4i_p)(gp + 8), (as3_v4i_p)(sp + 8), 0, 0);
}
#endif

// ---------------- fused denominator GEMMs (en + fr) -------------------------
// grid = (TOK/128, NSPLIT, 2); block = 256 (8 waves).
// Wave w owns m-tile blockIdx.x*8+w, keeps its 16x256 f16 A panel in VGPRs.
// All 8 waves share one candidate 16-row B tile, double-buffered in LDS.
__global__ void denom_gemm_kernel(const _Float16* __restrict__ zs_h,
                                  const _Float16* __restrict__ G_en,
                                  const _Float16* __restrict__ G_fr,
                                  const float* __restrict__ kap_en,
                                  const float* __restrict__ kap_fr,
                                  int P_en, int P_fr, int nc_en, int nc_fr,
                                  int nt_en, int nt_fr,
                                  float* __restrict__ den_en,
                                  float* __restrict__ den_fr) {
    __shared__ _Float16 ldsbuf[2][16 * D];   // 2 x 8KB, fragment-order B tiles

    const int lane = threadIdx.x & 31;
    const int wave = threadIdx.x >> 5;
    const int tid  = threadIdx.x;
    const int z    = blockIdx.z;

    const _Float16* G = z ? G_fr : G_en;
    const float kappa = z ? kap_fr[0] : kap_en[0];
    const int P       = z ? P_fr : P_en;
    const int Ncand   = z ? nc_fr : nc_en;
    const int ntiles  = z ? nt_fr : nt_en;
    float* denom      = z ? den_fr : den_en;

    const int mt    = blockIdx.x * 8 + wave;
    const int a_row = mt * 16 + (lane & 15);

    v16h a[8];
    #pragma unroll
    for (int ks = 0; ks < 8; ++ks) a[ks] = load_a_frag(zs_h, a_row, ks * 32, lane);

    float acc[8] = {0.f,0.f,0.f,0.f,0.f,0.f,0.f,0.f};

    // prologue: stage first tile into buffer 0
#if USE_ASYNC_LDS
    stage_async(G, blockIdx.y, tid, &ldsbuf[0][0]);
    __builtin_amdgcn_s_wait_asynccnt(0);
#else
    {
        uint4 s0, s1;
        stage_gload(G, blockIdx.y, tid, &s0, &s1);
        stage_sstore(&ldsbuf[0][0], tid, s0, s1);
    }
#endif
    __syncthreads();

    int p = 0;
    for (int nt = blockIdx.y; nt < ntiles; nt += NSPLIT) {
        const int next = nt + NSPLIT;
        const bool have_next = next < ntiles;     // block-uniform
#if USE_ASYNC_LDS
        if (have_next) stage_async(G, next, tid, &ldsbuf[p ^ 1][0]);
#else
        uint4 n0, n1;
        if (have_next) stage_gload(G, next, tid, &n0, &n1);
#endif

        // two-deep pipelined B fragments: ds_load(ks+1) in flight under wmma(ks)
        v8f c = {};
        v16h b_cur = load_b_lds(&ldsbuf[p][0], 0, lane);
        #pragma unroll
        for (int ks = 0; ks < 8; ++ks) {
            v16h b_next = b_cur;
            if (ks < 7) b_next = load_b_lds(&ldsbuf[p][0], ks + 1, lane);
            c = __builtin_amdgcn_wmma_f32_16x16x32_f16(false, a[ks], false, b_cur,
                                                       (short)0, c, false, false);
            b_cur = b_next;
        }

        const int col = nt * 16 + (lane & 15);
        const float w = (col < P) ? 1.0f : ((col < Ncand) ? kappa : 0.0f);
        #pragma unroll
        for (int r = 0; r < 8; ++r) acc[r] += w * __expf(c[r]);

#if USE_ASYNC_LDS
        if (have_next) __builtin_amdgcn_s_wait_asynccnt(0);
#else
        if (have_next) stage_sstore(&ldsbuf[p ^ 1][0], tid, n0, n1);
#endif
        __syncthreads();
        p ^= 1;
    }

    #pragma unroll
    for (int r = 0; r < 8; ++r) {
        float v = seg16_reduce(acc[r]);
        if ((lane & 15) == 0)
            atomicAdd(&denom[mt * 16 + r + ((lane >> 4) << 3)], v);
    }
}

// ---------------- en numerator logits: num[t] = z_t . W_en[x_en[t]] (fp32) --
__global__ void num_en_kernel(const float* __restrict__ zs,
                              const float* __restrict__ W,
                              const int* __restrict__ x,
                              float* __restrict__ num) {
    const int lane = threadIdx.x & 31;
    const int wave = threadIdx.x >> 5;
    const int t = blockIdx.x * 8 + wave;
    const float* z = zs + (size_t)t * D;
    const float* w = W + (size_t)x[t] * D;
    float p = 0.0f;
    #pragma unroll
    for (int i = 0; i < 8; ++i) { int d = lane + 32 * i; p += z[d] * w[d]; }
    p += __shfl_xor(p, 16, 32);
    p = seg16_reduce(p);
    if (lane == 0) num[t] = p;
}

// ---------------- fr alignment: out_fr[b] = sum_f log(sum_s e^{A_f.Z_s}/den_s)*mask
__global__ void fr_kernel(const _Float16* __restrict__ zs_h,
                          const _Float16* __restrict__ efr_h,
                          const float* __restrict__ denom_fr,
                          const float* __restrict__ fr_mask,
                          float* __restrict__ out_fr) {
    __shared__ float rowsum[SEQ];
    __shared__ float red[SEQ];
    const int lane = threadIdx.x & 31;
    const int wave = threadIdx.x >> 5;
    const int b = blockIdx.x;
    if (threadIdx.x < SEQ) rowsum[threadIdx.x] = 0.0f;
    __syncthreads();

    #pragma unroll
    for (int rep = 0; rep < 2; ++rep) {
        const int id = wave + rep * 8;          // 0..15
        const int ft = id >> 2, st = id & 3;    // 4x4 tiles of 16x16
        const int a_row = b * SEQ + ft * 16 + (lane & 15);   // f rows (W_fr[x_fr])
        const int b_row = b * SEQ + st * 16 + (lane & 15);   // s rows (zs)
        v8f c = {};
        #pragma unroll
        for (int kb = 0; kb < D; kb += 32) {
            v16h a = load_a_frag(efr_h, a_row, kb, lane);
            v16h bm = load_b_frag(zs_h, b_row, kb, lane);
            c = __builtin_amdgcn_wmma_f32_16x16x32_f16(false, a, false, bm,
                                                       (short)0, c, false, false);
        }
        const int s = st * 16 + (lane & 15);
        const float inv = 1.0f / denom_fr[b * SEQ + s];
        #pragma unroll
        for (int r = 0; r < 8; ++r) {
            float v = seg16_reduce(__expf(c[r]) * inv);
            if ((lane & 15) == 0)
                atomicAdd(&rowsum[ft * 16 + r + ((lane >> 4) << 3)], v);
        }
    }
    __syncthreads();
    if (threadIdx.x < SEQ)
        red[threadIdx.x] = __logf(rowsum[threadIdx.x]) * fr_mask[b * SEQ + threadIdx.x];
    __syncthreads();
    if (threadIdx.x == 0) {
        float s = 0.0f;
        for (int f = 0; f < SEQ; ++f) s += red[f];
        out_fr[b] = s;
    }
}

// ---------------- en loss: out_en[b] = sum_s (num - log(den))*mask ----------
__global__ void fin_en_kernel(const float* __restrict__ num,
                              const float* __restrict__ denom,
                              const float* __restrict__ mask,
                              float* __restrict__ out_en) {
    __shared__ float red[SEQ];
    const int b = blockIdx.x, s = threadIdx.x;
    const int t = b * SEQ + s;
    red[s] = (num[t] - __logf(denom[t])) * mask[t];
    __syncthreads();
    if (s == 0) {
        float acc = 0.0f;
        for (int i = 0; i < SEQ; ++i) acc += red[i];
        out_en[b] = acc;
    }
}

// ---------------- host launch ----------------------------------------------
static inline size_t align256(size_t x) { return (x + 255) & ~(size_t)255; }

extern "C" void kernel_launch(void* const* d_in, const int* in_sizes, int n_in,
                              void* d_out, int out_size, void* d_ws, size_t ws_size,
                              hipStream_t stream) {
    const float* zs      = (const float*)d_in[0];
    const int*   x_en    = (const int*)  d_in[1];
    const int*   x_fr    = (const int*)  d_in[2];
    const float* en_mask = (const float*)d_in[3];
    const float* fr_mask = (const float*)d_in[4];
    const float* W_en    = (const float*)d_in[5];
    const float* W_fr    = (const float*)d_in[6];
    const int*   pos_en  = (const int*)  d_in[7];
    const int*   neg_en  = (const int*)  d_in[8];
    const int*   pos_fr  = (const int*)  d_in[9];
    const int*   neg_fr  = (const int*)  d_in[10];
    const float* kap_en  = (const float*)d_in[11];
    const float* kap_fr  = (const float*)d_in[12];
    float* out = (float*)d_out;

    const int P_en = in_sizes[7], NEGn = in_sizes[8], P_fr = in_sizes[9];
    const int ncand_en = P_en + NEGn, ncand_fr = P_fr + NEGn;
    const int npad_en = (ncand_en + 15) & ~15;
    const int npad_fr = (ncand_fr + 15) & ~15;

    char* base = (char*)d_ws;
    size_t o = 0;
    _Float16* zs_h  = (_Float16*)(base + o); o = align256(o + (size_t)TOK * D * 2);
    _Float16* efr_h = (_Float16*)(base + o); o = align256(o + (size_t)TOK * D * 2);
    _Float16* G_en  = (_Float16*)(base + o); o = align256(o + (size_t)npad_en * D * 2);
    _Float16* G_fr  = (_Float16*)(base + o); o = align256(o + (size_t)npad_fr * D * 2);
    float* denom_en = (float*)(base + o);    o = align256(o + (size_t)TOK * 4);
    float* denom_fr = (float*)(base + o);    o = align256(o + (size_t)TOK * 4);
    float* num_en   = (float*)(base + o);    o = align256(o + (size_t)TOK * 4);
    (void)ws_size; (void)n_in; (void)out_size;

    zero_f32_kernel<<<(2 * TOK + 255) / 256, 256, 0, stream>>>(denom_en, 2 * TOK);
    cvt_f16_kernel<<<TOK, D, 0, stream>>>(zs, zs_h);
    gather_rows_f16_kernel<<<TOK, D, 0, stream>>>(W_fr, x_fr, efr_h);
    gather_cand_f16_kernel<<<npad_en, D, 0, stream>>>(W_en, pos_en, neg_en, P_en, ncand_en, G_en);
    gather_cand_f16_kernel<<<npad_fr, D, 0, stream>>>(W_fr, pos_fr, neg_fr, P_fr, ncand_fr, G_fr);

    dim3 dgrid(TOK / 128, NSPLIT, 2);
    denom_gemm_kernel<<<dgrid, 256, 0, stream>>>(zs_h, G_en, G_fr, kap_en, kap_fr,
                                                 P_en, P_fr, ncand_en, ncand_fr,
                                                 npad_en / 16, npad_fr / 16,
                                                 denom_en, denom_fr);

    num_en_kernel<<<TOK / 8, 256, 0, stream>>>(zs, W_en, x_en, num_en);
    fr_kernel<<<BATCH, 256, 0, stream>>>(zs_h, efr_h, denom_fr, fr_mask, out + BATCH);
    fin_en_kernel<<<BATCH, SEQ, 0, stream>>>(num_en, denom_en, en_mask, out);
}
